// AMI_38130719653974
// MI455X (gfx1250) — matrix operand order repaired and checked by reading, not compiled, not run on hardware
//
#include <hip/hip_runtime.h>
#include <cstdint>
#include <cstddef>

// Problem constants (match reference).
#define QN 4096
#define CN 4096
#define DN 1024
#define EPSF 1e-6f
#define ELEMS ((size_t)QN * DN)   // elements per matrix (Q==C here)

typedef __attribute__((ext_vector_type(16))) __bf16        v16bf;
typedef __attribute__((ext_vector_type(8)))  float         v8f;
typedef __attribute__((ext_vector_type(8)))  unsigned int  v8u;

struct U8 { uint4 a, b; };   // 32B carrier for one 8-VGPR bf16 fragment

__device__ __forceinline__ unsigned int pack2bf(float x0, float x1) {
  return (__float_as_uint(x0) >> 16) | (__float_as_uint(x1) & 0xFFFF0000u);
}

// ===========================================================================
// FAST PATH (ws >= 64MB+64KB): pre-split fp32 -> bf16 hi/lo into d_ws, then a
// load-b128 + WMMA GEMM with fused distance/exp epilogue.
// ===========================================================================

// ---------------------------------------------------------------------------
// Kernel P: one block per (row, matrix). Computes squared row norm AND writes
// row-major bf16 hi/lo split of the matrix.
//   sel 0: target_rgb (+eps), 1: target_flow (+eps), 2: context_rgb,
//   3: context_flow.  hi at pk + (2*sel)*ELEMS, lo at pk + (2*sel+1)*ELEMS.
// ---------------------------------------------------------------------------
__global__ __launch_bounds__(256) void prep_split_norms(
    const float* __restrict__ trgb, const float* __restrict__ tflow,
    const float* __restrict__ crgb, const float* __restrict__ cflow,
    float* __restrict__ norms, unsigned short* __restrict__ pk) {
  __shared__ float red[256];
  const int row = blockIdx.x;
  const int sel = blockIdx.y;
  const float* src = (sel == 0) ? trgb : (sel == 1) ? tflow
                   : (sel == 2) ? crgb : cflow;
  const size_t idx = (size_t)row * DN + threadIdx.x * 4;
  float4 x = *reinterpret_cast<const float4*>(src + idx);
  if (sel < 2) { x.x += EPSF; x.y += EPSF; x.z += EPSF; x.w += EPSF; }

  // hi = truncate-to-bf16, lo = bf16(x - hi)
  float h0 = __uint_as_float(__float_as_uint(x.x) & 0xFFFF0000u);
  float h1 = __uint_as_float(__float_as_uint(x.y) & 0xFFFF0000u);
  float h2 = __uint_as_float(__float_as_uint(x.z) & 0xFFFF0000u);
  float h3 = __uint_as_float(__float_as_uint(x.w) & 0xFFFF0000u);
  uint2 hi, lo;
  hi.x = pack2bf(x.x, x.y);        hi.y = pack2bf(x.z, x.w);
  lo.x = pack2bf(x.x - h0, x.y - h1);
  lo.y = pack2bf(x.z - h2, x.w - h3);
  *reinterpret_cast<uint2*>(pk + (size_t)(2 * sel)     * ELEMS + idx) = hi;
  *reinterpret_cast<uint2*>(pk + (size_t)(2 * sel + 1) * ELEMS + idx) = lo;

  red[threadIdx.x] = x.x * x.x + x.y * x.y + x.z * x.z + x.w * x.w;
  __syncthreads();
  for (int off = 128; off > 0; off >>= 1) {
    if ((int)threadIdx.x < off) red[threadIdx.x] += red[threadIdx.x + off];
    __syncthreads();
  }
  if (threadIdx.x == 0) norms[sel * 4096 + row] = red[0];
}

// ---------------------------------------------------------------------------
// Kernel G (fast): GEMM on prepacked bf16 hi/lo. 256 threads = 8 wave32s
// arranged 2(M) x 4(N); wave tile 64x64 (4x4 accumulator tiles), block tile
// 128x256, K stepped by 32. 48 WMMA per K-chunk vs 32 b128 loads.
// A-fragment (16x32 bf16, ISA 7.12.2): lane {half,m}; k = 8*half+0..7 in
// v0-3 and 16+8*half+0..7 in v4-7 -> two contiguous 16B loads.
// B-fragment (32x16): lane {half,n}; same k order as A -> two 16B loads.
// ---------------------------------------------------------------------------
template<bool ACCUM>
__global__ __launch_bounds__(256) void gemm_packed(
    const unsigned short* __restrict__ Ah, const unsigned short* __restrict__ Al,
    const unsigned short* __restrict__ Bh, const unsigned short* __restrict__ Bl,
    const float* __restrict__ anorm, const float* __restrict__ bnorm,
    const float* __restrict__ cnum,  const float* __restrict__ coth,
    float* __restrict__ out) {
  const int lane  = threadIdx.x & 31;
  const int wave  = threadIdx.x >> 5;
  const int waveM = wave & 1;   // 2 waves along M (64 rows each)
  const int waveN = wave >> 1;  // 4 waves along N (64 cols each)
  const int half  = lane >> 4;
  const int lrow  = lane & 15;

  const int q0 = blockIdx.y * 128 + waveM * 64;
  const int c0 = blockIdx.x * 256 + waveN * 64;

  size_t aoff[4], boff[4];
#pragma unroll
  for (int mt = 0; mt < 4; ++mt) aoff[mt] = (size_t)(q0 + mt * 16 + lrow) * DN;
#pragma unroll
  for (int nt = 0; nt < 4; ++nt) boff[nt] = (size_t)(c0 + nt * 16 + lrow) * DN;

  v8f acc[4][4] = {};

#pragma unroll 1
  for (int kc = 0; kc < DN; kc += 32) {
    v16bf ah[4], al[4], bh[4], bl[4];
#pragma unroll
    for (int mt = 0; mt < 4; ++mt) {
      const size_t base = aoff[mt] + kc + 8 * half;
      U8 uh, ul;
      uh.a = *reinterpret_cast<const uint4*>(Ah + base);
      uh.b = *reinterpret_cast<const uint4*>(Ah + base + 16);
      ul.a = *reinterpret_cast<const uint4*>(Al + base);
      ul.b = *reinterpret_cast<const uint4*>(Al + base + 16);
      ah[mt] = __builtin_bit_cast(v16bf, uh);
      al[mt] = __builtin_bit_cast(v16bf, ul);
    }
#pragma unroll
    for (int nt = 0; nt < 4; ++nt) {
      const size_t base = boff[nt] + kc + 8 * half;
      U8 uh, ul;
      uh.a = *reinterpret_cast<const uint4*>(Bh + base);
      uh.b = *reinterpret_cast<const uint4*>(Bh + base + 16);
      ul.a = *reinterpret_cast<const uint4*>(Bl + base);
      ul.b = *reinterpret_cast<const uint4*>(Bl + base + 16);
      bh[nt] = __builtin_bit_cast(v16bf, uh);
      bl[nt] = __builtin_bit_cast(v16bf, ul);
    }
#pragma unroll
    for (int mt = 0; mt < 4; ++mt) {
#pragma unroll
      for (int nt = 0; nt < 4; ++nt) {
        acc[mt][nt] = __builtin_amdgcn_wmma_f32_16x16x32_bf16(
            false, ah[mt], false, bh[nt], (short)0, acc[mt][nt], false, false);
        acc[mt][nt] = __builtin_amdgcn_wmma_f32_16x16x32_bf16(
            false, ah[mt], false, bl[nt], (short)0, acc[mt][nt], false, false);
        acc[mt][nt] = __builtin_amdgcn_wmma_f32_16x16x32_bf16(
            false, al[mt], false, bh[nt], (short)0, acc[mt][nt], false, false);
      }
    }
  }

  // Epilogue. C/D layout: VGPR r -> (M = r + 8*half, N = lane&15).
#pragma unroll
  for (int mt = 0; mt < 4; ++mt) {
#pragma unroll
    for (int r = 0; r < 8; ++r) {
      const int row = q0 + mt * 16 + r + 8 * half;
      const float an = anorm[row];
      const float cn = cnum[row];
      const float co = coth[row];
      const float w  = cn / (cn + co);
#pragma unroll
      for (int nt = 0; nt < 4; ++nt) {
        const int col  = c0 + nt * 16 + lrow;
        const float sq = an + bnorm[col] - 2.0f * acc[mt][nt][r];
        const float d  = sqrtf(fmaxf(sq, 0.0f));
        const float v  = w * __expf(-d);
        float* o = out + (size_t)row * CN + col;
        if (ACCUM) *o = *o + v; else *o = v;
      }
    }
  }
}

// ===========================================================================
// FALLBACK PATH (small ws): inline fp32->bf16 split inside the GEMM.
// ===========================================================================

struct FragPair { v8u hi; v8u lo; };

template<bool ADD_EPS, bool BMAP>
__device__ __forceinline__ FragPair load_frag(const float* __restrict__ rowp,
                                              int kbase, int half) {
  FragPair f;
#pragma unroll
  for (int v = 0; v < 8; ++v) {
    const int k = kbase + (BMAP ? (((v >> 2) << 4) + (half << 3) + ((v & 3) << 1))
                                : (((v >> 2) << 4) + (half << 3) + ((v & 3) << 1)));
    float2 x = *reinterpret_cast<const float2*>(rowp + k);
    float x0 = ADD_EPS ? (x.x + EPSF) : x.x;
    float x1 = ADD_EPS ? (x.y + EPSF) : x.y;
    float h0 = __uint_as_float(__float_as_uint(x0) & 0xFFFF0000u);
    float h1 = __uint_as_float(__float_as_uint(x1) & 0xFFFF0000u);
    f.hi[v] = pack2bf(h0, h1);
    f.lo[v] = pack2bf(x0 - h0, x1 - h1);
  }
  return f;
}

__global__ __launch_bounds__(256) void row_sq_norms(
    const float* __restrict__ trgb, const float* __restrict__ tflow,
    const float* __restrict__ crgb, const float* __restrict__ cflow,
    float* __restrict__ norms) {
  __shared__ float red[256];
  const int row = blockIdx.x;
  const int sel = blockIdx.y;
  const float* src = (sel == 0) ? trgb : (sel == 1) ? tflow
                   : (sel == 2) ? crgb : cflow;
  float4 x = *reinterpret_cast<const float4*>(src + (size_t)row * DN + threadIdx.x * 4);
  if (sel < 2) { x.x += EPSF; x.y += EPSF; x.z += EPSF; x.w += EPSF; }
  red[threadIdx.x] = x.x * x.x + x.y * x.y + x.z * x.z + x.w * x.w;
  __syncthreads();
  for (int off = 128; off > 0; off >>= 1) {
    if ((int)threadIdx.x < off) red[threadIdx.x] += red[threadIdx.x + off];
    __syncthreads();
  }
  if (threadIdx.x == 0) norms[sel * 4096 + row] = red[0];
}

template<bool ACCUM>
__global__ __launch_bounds__(256) void gemm_inline(
    const float* __restrict__ A, const float* __restrict__ B,
    const float* __restrict__ anorm, const float* __restrict__ bnorm,
    const float* __restrict__ cnum,  const float* __restrict__ coth,
    float* __restrict__ out) {
  const int lane  = threadIdx.x & 31;
  const int wave  = threadIdx.x >> 5;
  const int waveM = wave & 1;
  const int waveN = wave >> 1;
  const int half  = lane >> 4;
  const int lrow  = lane & 15;

  const int q0 = blockIdx.y * 128 + waveM * 64;
  const int c0 = blockIdx.x * 128 + waveN * 32;

  const float* arow[4];
  const float* brow[2];
#pragma unroll
  for (int mt = 0; mt < 4; ++mt) arow[mt] = A + (size_t)(q0 + mt * 16 + lrow) * DN;
#pragma unroll
  for (int nt = 0; nt < 2; ++nt) brow[nt] = B + (size_t)(c0 + nt * 16 + lrow) * DN;

  v8f acc[4][2] = {};

#pragma unroll 1
  for (int kc = 0; kc < DN; kc += 32) {
    FragPair a[4], b[2];
#pragma unroll
    for (int mt = 0; mt < 4; ++mt) a[mt] = load_frag<true,  false>(arow[mt], kc, half);
#pragma unroll
    for (int nt = 0; nt < 2; ++nt) b[nt] = load_frag<false, true >(brow[nt], kc, half);
#pragma unroll
    for (int mt = 0; mt < 4; ++mt) {
      const v16bf ah = __builtin_bit_cast(v16bf, a[mt].hi);
      const v16bf al = __builtin_bit_cast(v16bf, a[mt].lo);
#pragma unroll
      for (int nt = 0; nt < 2; ++nt) {
        const v16bf bh = __builtin_bit_cast(v16bf, b[nt].hi);
        const v16bf bl = __builtin_bit_cast(v16bf, b[nt].lo);
        acc[mt][nt] = __builtin_amdgcn_wmma_f32_16x16x32_bf16(
            false, ah, false, bh, (short)0, acc[mt][nt], false, false);
        acc[mt][nt] = __builtin_amdgcn_wmma_f32_16x16x32_bf16(
            false, ah, false, bl, (short)0, acc[mt][nt], false, false);
        acc[mt][nt] = __builtin_amdgcn_wmma_f32_16x16x32_bf16(
            false, al, false, bh, (short)0, acc[mt][nt], false, false);
      }
    }
  }

#pragma unroll
  for (int mt = 0; mt < 4; ++mt) {
#pragma unroll
    for (int r = 0; r < 8; ++r) {
      const int row = q0 + mt * 16 + r + 8 * half;
      const float an = anorm[row];
      const float cn = cnum[row];
      const float co = coth[row];
      const float w  = cn / (cn + co);
#pragma unroll
      for (int nt = 0; nt < 2; ++nt) {
        const int col  = c0 + nt * 16 + lrow;
        const float sq = an + bnorm[col] - 2.0f * acc[mt][nt][r];
        const float d  = sqrtf(fmaxf(sq, 0.0f));
        const float v  = w * __expf(-d);
        float* o = out + (size_t)row * CN + col;
        if (ACCUM) *o = *o + v; else *o = v;
      }
    }
  }
}

// ---------------------------------------------------------------------------
// Kernel N: in-place row normalization of the [Q x C] posterior.
// ---------------------------------------------------------------------------
__global__ __launch_bounds__(256) void normalize_rows(float* __restrict__ out) {
  __shared__ float red[256];
  const size_t base = (size_t)blockIdx.x * CN;
  float s = 0.0f;
  for (int i = threadIdx.x * 4; i < CN; i += 256 * 4) {
    float4 x = *reinterpret_cast<const float4*>(out + base + i);
    s += x.x + x.y + x.z + x.w;
  }
  red[threadIdx.x] = s;
  __syncthreads();
  for (int off = 128; off > 0; off >>= 1) {
    if ((int)threadIdx.x < off) red[threadIdx.x] += red[threadIdx.x + off];
    __syncthreads();
  }
  const float inv = 1.0f / red[0];
  for (int i = threadIdx.x * 4; i < CN; i += 256 * 4) {
    float4 x = *reinterpret_cast<const float4*>(out + base + i);
    x.x *= inv; x.y *= inv; x.z *= inv; x.w *= inv;
    *reinterpret_cast<float4*>(out + base + i) = x;
  }
}

// ---------------------------------------------------------------------------
extern "C" void kernel_launch(void* const* d_in, const int* in_sizes, int n_in,
                              void* d_out, int out_size, void* d_ws, size_t ws_size,
                              hipStream_t stream) {
  (void)in_sizes; (void)n_in; (void)out_size;
  const float* crgb  = (const float*)d_in[0];  // context_rgb_features  [C, D]
  const float* cflow = (const float*)d_in[1];  // context_flow_features [C, D]
  const float* trgb  = (const float*)d_in[2];  // target_rgb_features   [Q, D]
  const float* tflow = (const float*)d_in[3];  // target_flow_features  [Q, D]
  const float* c_r   = (const float*)d_in[4];  // [Q]
  const float* c_f   = (const float*)d_in[5];  // [Q]
  float* out   = (float*)d_out;                // [Q, C]
  float* norms = (float*)d_ws;                 // 4 * 4096 floats = 64 KB

  const size_t norms_bytes = 4 * 4096 * sizeof(float);
  const size_t pk_bytes    = 8 * ELEMS * sizeof(unsigned short);  // 64 MB

  if (ws_size >= norms_bytes + pk_bytes) {
    // Fast path: pre-split bf16 hi/lo, then pure load+WMMA GEMMs.
    unsigned short* pk = (unsigned short*)((char*)d_ws + norms_bytes);
    prep_split_norms<<<dim3(4096, 4), 256, 0, stream>>>(trgb, tflow, crgb, cflow,
                                                        norms, pk);
    dim3 grid(CN / 256, QN / 128);
    gemm_packed<false><<<grid, 256, 0, stream>>>(
        pk + 0 * ELEMS, pk + 1 * ELEMS,      // target_rgb hi/lo
        pk + 4 * ELEMS, pk + 5 * ELEMS,      // context_rgb hi/lo
        norms + 0, norms + 8192, c_r, c_f, out);
    gemm_packed<true><<<grid, 256, 0, stream>>>(
        pk + 2 * ELEMS, pk + 3 * ELEMS,      // target_flow hi/lo
        pk + 6 * ELEMS, pk + 7 * ELEMS,      // context_flow hi/lo
        norms + 4096, norms + 12288, c_f, c_r, out);
  } else {
    // Fallback: inline conversion GEMM (only needs 64 KB of ws).
    row_sq_norms<<<dim3(4096, 4), 256, 0, stream>>>(trgb, tflow, crgb, cflow, norms);
    dim3 grid(CN / 128, QN / 128);
    gemm_inline<false><<<grid, 256, 0, stream>>>(trgb, crgb, norms + 0, norms + 8192,
                                                 c_r, c_f, out);
    gemm_inline<true><<<grid, 256, 0, stream>>>(tflow, cflow, norms + 4096,
                                                norms + 12288, c_f, c_r, out);
  }
  normalize_rows<<<QN, 256, 0, stream>>>(out);
}
